// CascadedGroupAttention_37598143709589
// MI455X (gfx1250) — matrix operand order repaired and checked by reading, not compiled, not run on hardware
//
#include <hip/hip_runtime.h>
#include <cstdint>
#include <cstddef>

// ---------------- CDNA5 WMMA types ----------------
typedef _Float16 h16;
typedef _Float16 v16h __attribute__((ext_vector_type(16)));
typedef float    v8f  __attribute__((ext_vector_type(8)));

#define RESN   28
#define HW     784
#define BATCH  64
#define DIMC   512
#define C4     128          // q/k channels
#define HEADS  8
#define DHEAD  128          // per-head value dim
#define CIN_H  64           // per-head input channels
#define CAT    1024         // HEADS*DHEAD
#define EPSV   1e-5f
#define SCALEV 0.17677669529663687f   // 32^-0.5
#define LDSP   1026         // attn-apply LDS pitch (h16): 513 dwords -> +1 bank per K row
#define PQK    116          // qk f32 tile pitch (52-dword row step, 16 rows distinct banks)
#define PVS    120          // vs f16 tile pitch (60-dword row step, distinct banks)
#define PPJ    72           // proj transposed [n][k] pitch (contiguous k per lane)

// ---------------- WMMA fragment loaders (contiguous k order, same in A and B) ----------------
// A 16x32 f16 (MxK): lanes 0-15 hold M=lane, K=k0..k0+15 sequential;
// lanes 16-31 hold M=lane-16, K=k0+16..k0+31.
__device__ __forceinline__ v16h wmma_load_a(const h16* A, int lda, int m0, int k0, int lane) {
  int row = m0 + (lane & 15);
  int kb  = k0 + ((lane >> 4) << 4);
  const h16* p = A + row * lda + kb;
  v16h a;
#pragma unroll
  for (int i = 0; i < 8; ++i) { a[i] = p[i]; a[i + 8] = p[i + 8]; }
  return a;
}

// B 32x16 f16 (KxN): lanes 0-15 hold N=lane, K=k0..k0+15 sequential; lanes
// 16-31 hold N=lane-16, K=k0+16..k0+31.
__device__ __forceinline__ v16h wmma_load_b_lds(const h16* tile, int n0, int lane, int kmax) {
  int col = n0 + (lane & 15);
  int kb  = (lane >> 4) << 4;
  const h16* lp = tile + kb * LDSP + col;
  v16h b;
#pragma unroll
  for (int i = 0; i < 16; ++i)
    b[i] = (kb + i < kmax) ? lp[i * LDSP] : (h16)0.f;   // LDS OOB reads return 0 anyway
  return b;
}

#define WMMA_F16(a, b, c) \
  __builtin_amdgcn_wmma_f32_16x16x32_f16(false, (a), false, (b), (short)0, (c), false, false)

// Async global->LDS 16B copy (CDNA5 ASYNCcnt path, ISA 08 §4).
__device__ __forceinline__ void async_copy_b128(const void* gsrc, void* ldst) {
  unsigned long long g = (unsigned long long)gsrc;
  unsigned l = (unsigned)(uintptr_t)ldst;   // generic->LDS offset = low 32 bits
  asm volatile("global_load_async_to_lds_b128 %0, %1, off" :: "v"(l), "v"(g) : "memory");
}
__device__ __forceinline__ void async_wait0() {
  asm volatile("s_wait_asynccnt 0x0" ::: "memory");
}
__device__ __forceinline__ void async_wait7() {   // in-order completion: <=7 left
  asm volatile("s_wait_asynccnt 0x7" ::: "memory");
}

// ---------------- K0: fold BN into weights, build tables ----------------
__global__ void k_prep(const float* qk_w, const float* qk_bn, const float* dws_bn,
                       const float* vs_w, const float* vs_bn,
                       const float* pj_w, const float* pj_bn,
                       const float* bias_h, const float* bias_w, const int* bias_idx,
                       h16* w16_qk, float* b_qk, h16* w16_vs, float* b_vs,
                       h16* w16_pj, float* b_pj, float* dws_s, float* dws_t,
                       float* abw, float* abh) {
  int t = blockIdx.x * blockDim.x + threadIdx.x;
  if (t < DIMC * CAT) {                                      // proj weights 512x1024
    int o = t >> 10;
    float s = pj_bn[o] * rsqrtf(pj_bn[3 * DIMC + o] + EPSV);
    w16_pj[t] = (h16)(pj_w[t] * s);
  }
  if (t < 256 * DIMC) {                                      // qk weights 256x512
    int o = t >> 9;
    float s = qk_bn[o] * rsqrtf(qk_bn[3 * 256 + o] + EPSV);
    w16_qk[t] = (h16)(qk_w[t] * s);
  }
  if (t < HEADS * DHEAD * CIN_H) {                           // vs weights 8x128x64
    int i = t / (DHEAD * CIN_H);
    int d = (t / CIN_H) % DHEAD;
    const float* bn = vs_bn + i * 4 * DHEAD;
    float s = bn[d] * rsqrtf(bn[3 * DHEAD + d] + EPSV);
    w16_vs[t] = (h16)(vs_w[t] * s);
  }
  if (t < DIMC) {
    float s = pj_bn[t] * rsqrtf(pj_bn[3 * DIMC + t] + EPSV);
    b_pj[t] = pj_bn[DIMC + t] - pj_bn[2 * DIMC + t] * s;
  }
  if (t < 256) {
    float s = qk_bn[t] * rsqrtf(qk_bn[3 * 256 + t] + EPSV);
    b_qk[t] = qk_bn[256 + t] - qk_bn[2 * 256 + t] * s;
  }
  if (t < HEADS * DHEAD) {
    int i = t / DHEAD, d = t % DHEAD;
    const float* bn = vs_bn + i * 4 * DHEAD;
    float s = bn[d] * rsqrtf(bn[3 * DHEAD + d] + EPSV);
    b_vs[t] = bn[DHEAD + d] - bn[2 * DHEAD + d] * s;
  }
  if (t < C4) {
    float s = dws_bn[t] * rsqrtf(dws_bn[3 * C4 + t] + EPSV);
    dws_s[t] = s;
    dws_t[t] = dws_bn[C4 + t] - dws_bn[2 * C4 + t] * s;
  }
  if (t < HW) {                                              // rel-pos bias gather
    int idx = bias_idx[t];
    abw[t] = bias_w[idx];
    abh[t] = bias_h[idx];
  }
}

// ---------------- K1: qk 1x1 conv — 4 waves, async double-buffered LDS ---------
// Block computes 64(M) x 112(N); 16 K stages of 32, copy/compute overlapped.
__global__ void __launch_bounds__(128) k_qk_gemm(const h16* W, const float* x,
                                                 const float* bias,
                                                 float* q_pre, h16* k_t) {
  __shared__ float sB[2][32 * PQK];
  int tid = threadIdx.x, lane = tid & 31, wv = tid >> 5;
  int nbase = blockIdx.x * 112;
  int m0 = blockIdx.y * 64 + wv * 16;
  int b  = blockIdx.z;
  const float* X = x + (size_t)b * DIMC * HW + nbase;

  auto stage = [&](int ks, float* buf) {          // 32 rows x 112 f32 = 896 chunks
    const float* src = X + ks * 32 * HW;
    for (int c = tid; c < 896; c += 128) {        // 7 async issues per wave
      int r = c / 28, cc = (c % 28) * 4;
      async_copy_b128(src + r * HW + cc, &buf[r * PQK + cc]);
    }
  };

  v8f acc[7] = {};
  stage(0, sB[0]);
  for (int ks = 0; ks < 16; ++ks) {
    if (ks + 1 < 16) { stage(ks + 1, sB[(ks + 1) & 1]); async_wait7(); }
    else             { async_wait0(); }
    __syncthreads();
    const float* cur = sB[ks & 1];
    v16h a = wmma_load_a(W, DIMC, m0, ks * 32, lane);
    int col0 = lane & 15, kb = (lane >> 4) << 4;
#pragma unroll
    for (int j = 0; j < 7; ++j) {
      v16h bb;
#pragma unroll
      for (int i = 0; i < 16; ++i) bb[i] = (h16)cur[(kb + i) * PQK + j * 16 + col0];
      acc[j] = WMMA_F16(a, bb, acc[j]);
    }
    __syncthreads();
  }
  int mb = m0 + ((lane >> 4) << 3);
#pragma unroll
  for (int j = 0; j < 7; ++j) {
    int n = nbase + j * 16 + (lane & 15);
#pragma unroll
    for (int g = 0; g < 8; ++g) {
      int m = mb + g;
      float v = acc[j][g] + bias[m];
      if (m < C4) q_pre[(b * C4 + m) * HW + n] = v;             // q (pre-dws, NCHW f32)
      else        k_t[(b * HW + n) * C4 + (m - C4)] = (h16)v;   // k in [b,hw,c] f16
    }
  }
}

// ---------------- K2: depthwise 5x5 + BN, write q in [b,hw,c] f16 --------------
__global__ void k_dws(const float* q_pre, const float* dww, const float* sc,
                      const float* tb, h16* q_t) {
  int idx = blockIdx.x * blockDim.x + threadIdx.x;
  if (idx >= BATCH * C4 * HW) return;
  int hw = idx % HW;
  int c  = (idx / HW) % C4;
  int b  = idx / (HW * C4);
  int h = hw / RESN, w = hw % RESN;
  const float* src = q_pre + (b * C4 + c) * HW;
  const float* wk  = dww + c * 25;
  float acc = 0.f;
#pragma unroll
  for (int kh = 0; kh < 5; ++kh) {
    int ih = h + kh - 2;
    if (ih < 0 || ih >= RESN) continue;
#pragma unroll
    for (int kw = 0; kw < 5; ++kw) {
      int iw = w + kw - 2;
      if (iw < 0 || iw >= RESN) continue;
      acc += src[ih * RESN + iw] * wk[kh * 5 + kw];
    }
  }
  q_t[(b * HW + hw) * C4 + c] = (h16)(acc * sc[c] + tb[c]);
}

// ---------------- K3: head-chunk prefix sums of x (cascade linearized) ---------
__global__ void k_prefix(const float* x, h16* xacc) {
  int idx = blockIdx.x * blockDim.x + threadIdx.x;
  if (idx >= BATCH * CIN_H * HW) return;
  int hw = idx % HW;
  int c  = (idx / HW) % CIN_H;
  int b  = idx / (HW * CIN_H);
  float acc = 0.f;
#pragma unroll
  for (int i = 0; i < HEADS; ++i) {
    acc += x[((b * DIMC) + i * CIN_H + c) * HW + hw];
    xacc[((b * HEADS + i) * CIN_H + c) * HW + hw] = (h16)acc;
  }
}

// ---------------- K4: attention scores + softmax, zero-padded 32x32 f16 tiles --
__global__ void k_attn(const h16* q_t, const h16* k_t, const float* abw, const float* abh,
                       h16* attnW, h16* attnH) {
  __shared__ float qs[RESN * C4];
  __shared__ float ks[RESN * C4];
  __shared__ float sm[RESN * RESN];
  int line = blockIdx.x;
  int type = blockIdx.y;   // 0: row(w) attention at fixed h; 1: col(h) attention at fixed w
  int b    = blockIdx.z;
  int tid  = threadIdx.x;  // 256 threads
  for (int i = tid; i < RESN * C4; i += 256) {
    int p = i / C4, c = i % C4;
    int hw = (type == 0) ? line * RESN + p : p * RESN + line;
    int base = (b * HW + hw) * C4 + c;
    qs[i] = (float)q_t[base];
    ks[i] = (float)k_t[base];
  }
  __syncthreads();
  const float* ab = (type == 0) ? abw : abh;
  for (int i = tid; i < RESN * RESN; i += 256) {
    int qq = i / RESN, kk = i % RESN;
    const float* qp = qs + qq * C4;
    const float* kp = ks + kk * C4;
    float d = 0.f;
    for (int c = 0; c < C4; ++c) d += qp[c] * kp[c];
    sm[i] = d * SCALEV + ab[i];
  }
  __syncthreads();
  h16* out = ((type == 0) ? attnW : attnH) + (b * RESN + line) * 1024;
  for (int i = tid; i < 1024; i += 256) out[i] = (h16)0.f;   // zero 32x32 pad
  __syncthreads();
  if (tid < RESN) {
    float* row = sm + tid * RESN;
    float mx = row[0];
    for (int k = 1; k < RESN; ++k) mx = fmaxf(mx, row[k]);
    float s = 0.f;
    for (int k = 0; k < RESN; ++k) { row[k] = __expf(row[k] - mx); s += row[k]; }
    float inv = 1.f / s;
    for (int k = 0; k < RESN; ++k) out[tid * 32 + k] = (h16)(row[k] * inv);
  }
}

// ---------------- K5: per-head value conv — 4 waves, async LDS staged ----------
// Block computes 64(M=d) x 112(N); whole K=64 staged once.
__global__ void __launch_bounds__(128) k_vs_gemm(const h16* Wv, const h16* xacc,
                                                 const float* bias, h16* v_t) {
  __shared__ h16 sB[CIN_H * PVS];
  int tid = threadIdx.x, lane = tid & 31, wv = tid >> 5;
  int nbase = blockIdx.x * 112;
  int mt = blockIdx.y * 4 + wv;       // grid.y = 2 -> mt 0..7
  int bh = blockIdx.z;                // b*8 + head
  int head = bh & 7, b = bh >> 3;
  const h16* B = xacc + (size_t)bh * CIN_H * HW + nbase;
  for (int c = tid; c < 896; c += 128) {          // 64 rows x 112 h16 = 896 chunks
    int r = c / 14, cc = (c % 14) * 8;
    async_copy_b128(B + r * HW + cc, &sB[r * PVS + cc]);
  }
  async_wait0();
  __syncthreads();
  const h16* A = Wv + head * DHEAD * CIN_H;
  v8f acc[7] = {};
  int col0 = lane & 15;
#pragma unroll
  for (int ksub = 0; ksub < 2; ++ksub) {
    v16h a = wmma_load_a(A, CIN_H, mt * 16, ksub * 32, lane);
    int kb = ksub * 32 + ((lane >> 4) << 4);
#pragma unroll
    for (int j = 0; j < 7; ++j) {
      v16h bb;
#pragma unroll
      for (int i = 0; i < 16; ++i) bb[i] = sB[(kb + i) * PVS + j * 16 + col0];
      acc[j] = WMMA_F16(a, bb, acc[j]);
    }
  }
  int db = mt * 16 + ((lane >> 4) << 3);
  const float* bp = bias + head * DHEAD + db;
#pragma unroll
  for (int j = 0; j < 7; ++j) {
    int n = nbase + j * 16 + (lane & 15);
    h16* dst = v_t + (size_t)(b * HW + n) * CAT + head * DHEAD + db;
#pragma unroll
    for (int g = 0; g < 8; ++g) dst[g] = (h16)(acc[j][g] + bp[g]);
  }
}

// ---------------- K6: row attention apply — attn_w[32x32] x v[32x1024] ---------
__global__ void __launch_bounds__(128) k_row_attn(const h16* attnW, const h16* v_t,
                                                  h16* t_buf) {
  __shared__ h16 tile[RESN * LDSP];
  int tid = threadIdx.x, lane = tid & 31, wv = tid >> 5;
  int h = blockIdx.x, b = blockIdx.y;
  const h16* Bg = v_t + (size_t)(b * HW + h * RESN) * CAT;   // rows k=w, stride CAT
  for (int c = tid; c < RESN * 128; c += 128) {              // 16B chunks
    int r = c >> 7, cc = (c & 127) << 3;
    async_copy_b128(Bg + r * CAT + cc, &tile[r * LDSP + cc]);
  }
  async_wait0();
  __syncthreads();
  const h16* A = attnW + (b * RESN + h) * 1024;              // padded 32x32
  v16h a0 = wmma_load_a(A, 32, 0, 0, lane);
  v16h a1 = wmma_load_a(A, 32, 16, 0, lane);
  int q0 = (lane >> 4) << 3, nl = lane & 15;
  h16* dbase = t_buf + (size_t)(b * RESN + h) * RESN * CAT;
  for (int nt = wv * 16; nt < wv * 16 + 16; ++nt) {
    int n0 = nt * 16;
    v16h bb = wmma_load_b_lds(tile, n0, lane, RESN);
    v8f c0 = {}, c1 = {};
    c0 = WMMA_F16(a0, bb, c0);
    c1 = WMMA_F16(a1, bb, c1);
    h16* dst = dbase + n0 + nl;
#pragma unroll
    for (int g = 0; g < 8; ++g) {
      int q = q0 + g;                     // 0..15 always valid
      dst[q * CAT] = (h16)c0[g];
      int q2 = q + 16;
      if (q2 < RESN) dst[q2 * CAT] = (h16)c1[g];
    }
  }
}

// ---------------- K7: col attention apply + ReLU (writes f over v buffer) ------
__global__ void __launch_bounds__(128) k_col_attn(const h16* attnH, const h16* t_buf,
                                                  h16* f) {
  __shared__ h16 tile[RESN * LDSP];
  int tid = threadIdx.x, lane = tid & 31, wv = tid >> 5;
  int w = blockIdx.x, b = blockIdx.y;
  const h16* Bg = t_buf + (size_t)(b * HW + w) * CAT;        // rows k=h, stride RESN*CAT
  for (int c = tid; c < RESN * 128; c += 128) {
    int r = c >> 7, cc = (c & 127) << 3;
    async_copy_b128(Bg + (size_t)r * RESN * CAT + cc, &tile[r * LDSP + cc]);
  }
  async_wait0();
  __syncthreads();
  const h16* A = attnH + (b * RESN + w) * 1024;
  v16h a0 = wmma_load_a(A, 32, 0, 0, lane);
  v16h a1 = wmma_load_a(A, 32, 16, 0, lane);
  int q0 = (lane >> 4) << 3, nl = lane & 15;
  h16* dbase = f + (size_t)(b * HW + w * RESN) * CAT;
  for (int nt = wv * 16; nt < wv * 16 + 16; ++nt) {
    int n0 = nt * 16;
    v16h bb = wmma_load_b_lds(tile, n0, lane, RESN);
    v8f c0 = {}, c1 = {};
    c0 = WMMA_F16(a0, bb, c0);
    c1 = WMMA_F16(a1, bb, c1);
    h16* dst = dbase + n0 + nl;
#pragma unroll
    for (int g = 0; g < 8; ++g) {
      int q = q0 + g;
      dst[q * CAT] = (h16)fmaxf((float)c0[g], 0.f);          // ReLU fused
      int q2 = q + 16;
      if (q2 < RESN) dst[q2 * CAT] = (h16)fmaxf((float)c1[g], 0.f);
    }
  }
}

// ---------------- K8: projection — 4 waves, async double-buffered, [n][k] tile -
__global__ void __launch_bounds__(128) k_proj(const h16* Wp, const h16* f,
                                              const float* bias, float* out) {
  __shared__ h16 sB[2][112 * PPJ];    // transposed [n][k]: contiguous k per lane
  int tid = threadIdx.x, lane = tid & 31, wv = tid >> 5;
  int nbase = blockIdx.x * 112;
  int m0 = blockIdx.y * 64 + wv * 16;
  int b  = blockIdx.z;

  auto stage = [&](int ks, h16* buf) {          // 112 cols x 64 h16 = 896 chunks
    int k0 = ks * 64;
    for (int c = tid; c < 896; c += 128) {      // 7 async issues per wave
      int colc = c >> 3, cc = (c & 7) * 8;
      int n = nbase + colc;
      const h16* src = f + (size_t)((b * RESN + n % RESN) * RESN + n / RESN) * CAT + k0 + cc;
      async_copy_b128(src, &buf[colc * PPJ + cc]);
    }
  };

  v8f acc[7] = {};
  stage(0, sB[0]);
  for (int ks = 0; ks < 16; ++ks) {
    if (ks + 1 < 16) { stage(ks + 1, sB[(ks + 1) & 1]); async_wait7(); }
    else             { async_wait0(); }
    __syncthreads();
    const h16* cur = sB[ks & 1];
    int col0 = lane & 15;
#pragma unroll
    for (int ksub = 0; ksub < 2; ++ksub) {
      v16h a = wmma_load_a(Wp, CAT, m0, ks * 64 + ksub * 32, lane);
      int kb = ksub * 32 + ((lane >> 4) << 4);
#pragma unroll
      for (int j = 0; j < 7; ++j) {
        const h16* lp = cur + (j * 16 + col0) * PPJ + kb;
        v16h bb;
#pragma unroll
        for (int i = 0; i < 16; ++i) bb[i] = lp[i];
        acc[j] = WMMA_F16(a, bb, acc[j]);
      }
    }
    __syncthreads();
  }
  int mb = m0 + ((lane >> 4) << 3);
#pragma unroll
  for (int j = 0; j < 7; ++j) {
    int n = nbase + j * 16 + (lane & 15);
#pragma unroll
    for (int g = 0; g < 8; ++g) {
      int m = mb + g;
      out[((size_t)b * DIMC + m) * HW + n] = acc[j][g] + bias[m];
    }
  }
}

// ---------------- host-side launch ----------------
extern "C" void kernel_launch(void* const* d_in, const int* in_sizes, int n_in,
                              void* d_out, int out_size, void* d_ws, size_t ws_size,
                              hipStream_t stream) {
  const float* x       = (const float*)d_in[0];
  const float* qk_w    = (const float*)d_in[1];
  const float* qk_bn   = (const float*)d_in[2];
  const float* dws_w   = (const float*)d_in[3];
  const float* dws_bn  = (const float*)d_in[4];
  const float* vs_w    = (const float*)d_in[5];
  const float* vs_bn   = (const float*)d_in[6];
  const float* pj_w    = (const float*)d_in[7];
  const float* pj_bn   = (const float*)d_in[8];
  const float* bias_h  = (const float*)d_in[9];
  const float* bias_w  = (const float*)d_in[10];
  const int*   bias_ix = (const int*)d_in[11];
  float* out = (float*)d_out;

  // Carve workspace (256B aligned slots). Total ~320 MB.
  size_t off = 0;
  auto carve = [&](size_t bytes) -> void* {
    void* p = (char*)d_ws + off;
    off = (off + bytes + 255) & ~(size_t)255;
    return p;
  };
  h16*   w16_qk = (h16*)  carve((size_t)256 * DIMC * 2);
  float* b_qk   = (float*)carve(256 * 4);
  h16*   w16_vs = (h16*)  carve((size_t)HEADS * DHEAD * CIN_H * 2);
  float* b_vs   = (float*)carve(CAT * 4);
  h16*   w16_pj = (h16*)  carve((size_t)DIMC * CAT * 2);
  float* b_pj   = (float*)carve(DIMC * 4);
  float* dws_s  = (float*)carve(C4 * 4);
  float* dws_t  = (float*)carve(C4 * 4);
  float* abw    = (float*)carve(HW * 4);
  float* abh    = (float*)carve(HW * 4);
  float* q_pre  = (float*)carve((size_t)BATCH * C4 * HW * 4);
  h16*   q_t    = (h16*)  carve((size_t)BATCH * HW * C4 * 2);
  h16*   k_t    = (h16*)  carve((size_t)BATCH * HW * C4 * 2);
  h16*   xacc   = (h16*)  carve((size_t)BATCH * HEADS * CIN_H * HW * 2);
  h16*   attnW  = (h16*)  carve((size_t)BATCH * RESN * 1024 * 2);
  h16*   attnH  = (h16*)  carve((size_t)BATCH * RESN * 1024 * 2);
  h16*   v_t    = (h16*)  carve((size_t)BATCH * HW * CAT * 2);
  h16*   t_buf  = (h16*)  carve((size_t)BATCH * HW * CAT * 2);
  h16*   f_buf  = v_t;   // col-attn output reuses the v buffer (v is dead by then)
  (void)ws_size; (void)in_sizes; (void)n_in; (void)out_size;

  // 0) BN folding + tables
  k_prep<<<(DIMC * CAT + 255) / 256, 256, 0, stream>>>(
      qk_w, qk_bn, dws_bn, vs_w, vs_bn, pj_w, pj_bn, bias_h, bias_w, bias_ix,
      w16_qk, b_qk, w16_vs, b_vs, w16_pj, b_pj, dws_s, dws_t, abw, abh);

  // 1) qk 1x1 conv (64x112 per block, async pipelined)
  k_qk_gemm<<<dim3(7, 4, BATCH), 128, 0, stream>>>(w16_qk, x, b_qk, q_pre, k_t);

  // 2) depthwise 5x5 + BN on q
  k_dws<<<(BATCH * C4 * HW + 255) / 256, 256, 0, stream>>>(q_pre, dws_w, dws_s, dws_t, q_t);

  // 3) cascade prefix sums (heads are independent given prefixes)
  k_prefix<<<(BATCH * CIN_H * HW + 255) / 256, 256, 0, stream>>>(x, xacc);

  // 4) attention scores + softmax (both axes)
  k_attn<<<dim3(RESN, 2, BATCH), 256, 0, stream>>>(q_t, k_t, abw, abh, attnW, attnH);

  // 5) per-head value convs (64x112 per block, async staged)
  k_vs_gemm<<<dim3(7, 2, BATCH * HEADS), 128, 0, stream>>>(w16_vs, xacc, b_vs, v_t);

  // 6) row attention apply (all heads at once: N=1024), async LDS staging
  k_row_attn<<<dim3(RESN, BATCH), 128, 0, stream>>>(attnW, v_t, t_buf);

  // 7) col attention apply + ReLU, async LDS staging
  k_col_attn<<<dim3(RESN, BATCH), 128, 0, stream>>>(attnH, t_buf, f_buf);

  // 8) projection + BN -> output (64x112 per block, async pipelined)
  k_proj<<<dim3(7, 8, BATCH), 128, 0, stream>>>(w16_pj, f_buf, b_pj, out);
}
